// EnsembleHead_67611375174229
// MI455X (gfx1250) — matrix operand order repaired and checked by reading, not compiled, not run on hardware
//
#include <hip/hip_runtime.h>
#include <hip/hip_bf16.h>

// ---------------------------------------------------------------------------
// Types for CDNA5 WMMA (wave32): v_wmma_f32_16x16x32_bf16
// ---------------------------------------------------------------------------
typedef __bf16 v16bf __attribute__((ext_vector_type(16)));
typedef float  v8f   __attribute__((ext_vector_type(8)));

union FragU { v16bf v; uint4 q[2]; };

static __device__ __forceinline__ unsigned short f32_to_bf16(float f) {
    unsigned u = __float_as_uint(f);
    unsigned r = (u + 0x7FFFu + ((u >> 16) & 1u)) >> 16;   // round-to-nearest-even
    return (unsigned short)r;
}
static __device__ __forceinline__ float bf16_to_f32(unsigned short h) {
    return __uint_as_float(((unsigned)h) << 16);
}

// A-matrix fragment (16x32 bf16, ISA 7.12.2): lane l holds row (l&15),
// K runs [half*8, half*8+8) and [16+half*8, ...) where half = l>>4.
static __device__ __forceinline__ v16bf load_frag_a(const unsigned short* base,
                                                    int row_stride, int row0, int k0) {
    int lane = threadIdx.x & 31;
    int r = row0 + (lane & 15);
    int half = lane >> 4;
    const unsigned short* p = base + (size_t)r * row_stride + k0 + half * 8;
    FragU f;
    f.q[0] = *(const uint4*)(p);
    f.q[1] = *(const uint4*)(p + 16);
    return f.v;
}

// B-matrix fragment (32x16 bf16): lane l holds column (l&15), K runs
// [half*8, half*8+8) and [16+half*8, ...). Source stored row-major as W[N][K], so
// column n of B == row n of W.
static __device__ __forceinline__ v16bf load_frag_b(const unsigned short* base,
                                                    int row_stride, int n0, int k0) {
    int lane = threadIdx.x & 31;
    int n = n0 + (lane & 15);
    int half = lane >> 4;
    const unsigned short* p = base + (size_t)n * row_stride + k0 + half * 8;
    FragU f;
    f.q[0] = *(const uint4*)(p);
    f.q[1] = *(const uint4*)(p + 16);
    return f.v;
}

static __device__ __forceinline__ float sigmoidf_fast(float x) {
    return 1.0f / (1.0f + __expf(-x));
}

#define B_   1024
#define N_   512
#define DIN_ 30
#define H_   64
#define G4_  256     // 4*H
#define M_   (B_ * N_)   // 524288 flattened rows (b*512 + t)

// ---------------------------------------------------------------------------
// Kernel 1: fold W_fc into W_ih, build bf16 weights. One block of 256 threads.
//   Wcomb[n][k] (256 x 32 bf16, k>=30 zero), bcomb[n] = W_ih@b_fc + b_ih + b_hh
//   Whh_bf[n][k] (256 x 64 bf16)
// ---------------------------------------------------------------------------
__global__ __launch_bounds__(256) void prep_kernel(
    const float* __restrict__ W_fc, const float* __restrict__ b_fc,
    const float* __restrict__ W_ih, const float* __restrict__ W_hh,
    const float* __restrict__ b_ih, const float* __restrict__ b_hh,
    unsigned short* __restrict__ Wcomb, float* __restrict__ bcomb,
    unsigned short* __restrict__ Whh_bf)
{
    int n = threadIdx.x;           // 0..255 (gate row)
    float acc[DIN_];
    #pragma unroll
    for (int j = 0; j < DIN_; j++) acc[j] = 0.f;
    float bacc = 0.f;
    for (int c = 0; c < H_; c++) {
        float w = W_ih[n * H_ + c];
        bacc += w * b_fc[c];
        #pragma unroll
        for (int j = 0; j < DIN_; j++) acc[j] += w * W_fc[c * DIN_ + j];
    }
    #pragma unroll
    for (int j = 0; j < DIN_; j++) Wcomb[n * 32 + j] = f32_to_bf16(acc[j]);
    Wcomb[n * 32 + 30] = 0; Wcomb[n * 32 + 31] = 0;
    bcomb[n] = bacc + b_ih[n] + b_hh[n];
    #pragma unroll
    for (int k = 0; k < H_; k++) Whh_bf[n * H_ + k] = f32_to_bf16(W_hh[n * H_ + k]);
}

// ---------------------------------------------------------------------------
// Kernel 2: xg[m][g] = x[m][0:30] @ Wcomb.T + bcomb   (bf16 out, f32 accum)
//   M = 524288, N = 256, K = 32. 128 rows / block, 8 waves, 16 N-tiles each.
// ---------------------------------------------------------------------------
#define GB_ROWS 128
__global__ __launch_bounds__(256) void gemm1_kernel(
    const float* __restrict__ x, const unsigned short* __restrict__ Wcomb,
    const float* __restrict__ bcomb, unsigned short* __restrict__ xg)
{
    __shared__ __align__(16) unsigned short sW[G4_ * 32];     // 16 KB
    __shared__ __align__(16) unsigned short sX[GB_ROWS * 32]; //  8 KB
    __shared__ float sB[G4_];

    int tid = threadIdx.x;
    size_t m0 = (size_t)blockIdx.x * GB_ROWS;

    // stage Wcomb (8192 halfwords) via 16B chunks
    {
        const uint4* src = (const uint4*)Wcomb;
        uint4* dst = (uint4*)sW;
        #pragma unroll
        for (int i = tid; i < (G4_ * 32) / 8; i += 256) dst[i] = src[i];
    }
    sB[tid] = bcomb[tid];

    // stage + convert x block: 2 threads per row, 15 floats each; pad K 30..31
    {
        int r = tid >> 1, hf = tid & 1;
        const float* xr = x + (m0 + r) * DIN_ + hf * 15;
        unsigned short* dr = sX + r * 32 + hf * 15;
        #pragma unroll
        for (int j = 0; j < 15; j++) dr[j] = f32_to_bf16(xr[j]);
        if (hf) { sX[r * 32 + 30] = 0; sX[r * 32 + 31] = 0; }
    }
    __syncthreads();

    int wave = tid >> 5;
    int lane = tid & 31;
    int half = lane >> 4;

    v16bf a = load_frag_a(sX, 32, wave * 16, 0);

    for (int nt = 0; nt < 16; nt++) {
        v16bf b = load_frag_b(sW, 32, nt * 16, 0);
        int n = nt * 16 + (lane & 15);
        float bias = sB[n];
        v8f c;
        #pragma unroll
        for (int v = 0; v < 8; v++) c[v] = bias;
        c = __builtin_amdgcn_wmma_f32_16x16x32_bf16(false, a, false, b,
                                                    (short)0, c, false, false);
        unsigned short* outp = xg + (m0 + wave * 16 + 8 * half) * G4_ + n;
        #pragma unroll
        for (int v = 0; v < 8; v++) outp[(size_t)v * G4_] = f32_to_bf16(c[v]);
    }
}

// ---------------------------------------------------------------------------
// Kernel 3: LSTM scan. 64 blocks x 128 threads (4 waves); 16 batch rows/block.
//   W_hh fragments persistent in VGPRs (4 N-tiles x 2 K-frags per wave),
//   h in LDS (bf16), c in registers, logit = h @ W_last fused via ds_add_f32.
//   xg slice streamed with GLOBAL_LOAD_ASYNC_TO_LDS_B128 (ASYNCcnt), so the
//   8 KB/step DMA overlaps the WMMA recurrence; fenced with s_wait_asynccnt.
// ---------------------------------------------------------------------------
__global__ __launch_bounds__(128) void lstm_kernel(
    const unsigned short* __restrict__ xg, const unsigned short* __restrict__ Whh_bf,
    const float* __restrict__ W_last, const float* __restrict__ b_last,
    float* __restrict__ logits)
{
    __shared__ __align__(16) unsigned short sH[16 * H_];    //  2 KB  (h, bf16)
    __shared__ float          sG[16 * G4_];                 // 16 KB  (gate preacts)
    __shared__ __align__(16) unsigned short sXG[16 * G4_];  //  8 KB  (xg slice)
    __shared__ float          sLogit[16];

    int tid  = threadIdx.x;
    int wave = tid >> 5;
    int lane = tid & 31;
    int half = lane >> 4;
    int b0   = blockIdx.x * 16;

    // persistent B fragments of W_hh^T: wave owns N-tiles 4w..4w+3, K = 2x32
    v16bf Bf[4][2];
    #pragma unroll
    for (int j = 0; j < 4; j++) {
        int nt = wave * 4 + j;
        #pragma unroll
        for (int kf = 0; kf < 2; kf++)
            Bf[j][kf] = load_frag_b(Whh_bf, H_, nt * 16, kf * 32);
    }

    for (int i = tid; i < 16 * H_; i += 128) sH[i] = 0;
    if (tid < 16) sLogit[tid] = 0.f;

    // per-thread cell ownership: row r (16 rows), 8 hidden units each
    int r  = tid >> 3;
    int c0 = (tid & 7) * 8;
    float cst[8], wl[8];
    #pragma unroll
    for (int v = 0; v < 8; v++) { cst[v] = 0.f; wl[v] = W_last[c0 + v]; }
    float blast = b_last[0];

    const unsigned short* xgb = xg + (size_t)b0 * N_ * G4_;
    __syncthreads();

    #pragma unroll 1
    for (int t = 0; t < N_; t++) {
        // async-DMA xg[t] slice (16 rows x 512 B) straight into LDS; the DMA
        // runs while this wave does its WMMAs. Also prefetch t+1 (+512 B).
        #pragma unroll
        for (int i = tid; i < 512; i += 128) {
            int rr = i >> 5, cc = i & 31;            // 32 x 16B chunks per row
            const uint4* src = (const uint4*)(xgb + ((size_t)rr * N_ + t) * G4_) + cc;
            unsigned ldsoff = (unsigned)(size_t)((uint4*)sXG + rr * 32 + cc);
            asm volatile("global_load_async_to_lds_b128 %0, %1, off"
                         :: "v"(ldsoff), "v"(src) : "memory");
            __builtin_prefetch((const void*)(src + 32), 0, 0);  // next t: +512 B
        }

        // h @ W_hh^T  (A from sH; D -> sG)
        v16bf A0 = load_frag_a(sH, H_, 0, 0);
        v16bf A1 = load_frag_a(sH, H_, 0, 32);
        #pragma unroll
        for (int j = 0; j < 4; j++) {
            v8f c;
            #pragma unroll
            for (int v = 0; v < 8; v++) c[v] = 0.f;
            c = __builtin_amdgcn_wmma_f32_16x16x32_bf16(false, A0, false, Bf[j][0],
                                                        (short)0, c, false, false);
            c = __builtin_amdgcn_wmma_f32_16x16x32_bf16(false, A1, false, Bf[j][1],
                                                        (short)0, c, false, false);
            int n = (wave * 4 + j) * 16 + (lane & 15);
            #pragma unroll
            for (int v = 0; v < 8; v++) sG[(v + 8 * half) * G4_ + n] = c[v];
        }
        // make the async DMA visible before the consuming barrier
        asm volatile("s_wait_asynccnt 0x0" ::: "memory");
        __syncthreads();

        // gate nonlinearities + cell update + fused logit partial
        float lpart = 0.f, hrow[8];
        #pragma unroll
        for (int v = 0; v < 8; v++) {
            int cc = c0 + v;
            const float* gr = sG + r * G4_;
            const unsigned short* xr = sXG + r * G4_;
            float gi = sigmoidf_fast(gr[cc]       + bf16_to_f32(xr[cc]));
            float gf = sigmoidf_fast(gr[cc +  64] + bf16_to_f32(xr[cc +  64]));
            float gg = tanhf        (gr[cc + 128] + bf16_to_f32(xr[cc + 128]));
            float go = sigmoidf_fast(gr[cc + 192] + bf16_to_f32(xr[cc + 192]));
            cst[v] = gf * cst[v] + gi * gg;
            float h = go * tanhf(cst[v]);
            hrow[v] = h;
            lpart += h * wl[v];
        }
        #pragma unroll
        for (int v = 0; v < 8; v++) sH[r * H_ + c0 + v] = f32_to_bf16(hrow[v]);
        atomicAdd(&sLogit[r], lpart);                 // ds_add_f32
        __syncthreads();

        if (tid < 16) {
            logits[(size_t)(b0 + tid) * N_ + t] = sLogit[tid] + blast;
            sLogit[tid] = 0.f;   // ordered before next adds by next iter's barrier
        }
    }
}

// ---------------------------------------------------------------------------
// Kernel 4: softmax over N per batch row. 1024 blocks x 256 threads.
// ---------------------------------------------------------------------------
__global__ __launch_bounds__(256) void softmax_kernel(
    const float* __restrict__ logits, float* __restrict__ out)
{
    int b = blockIdx.x, tid = threadIdx.x;
    const float* row = logits + (size_t)b * N_;
    float a0 = row[tid], a1 = row[tid + 256];

    float m = fmaxf(a0, a1);
    #pragma unroll
    for (int off = 16; off; off >>= 1) m = fmaxf(m, __shfl_xor(m, off, 32));
    __shared__ float redM[8], redS[8];
    if ((tid & 31) == 0) redM[tid >> 5] = m;
    __syncthreads();
    float mm = redM[0];
    #pragma unroll
    for (int i = 1; i < 8; i++) mm = fmaxf(mm, redM[i]);

    float e0 = __expf(a0 - mm), e1 = __expf(a1 - mm);
    float s = e0 + e1;
    #pragma unroll
    for (int off = 16; off; off >>= 1) s += __shfl_xor(s, off, 32);
    if ((tid & 31) == 0) redS[tid >> 5] = s;
    __syncthreads();
    float ss = 0.f;
    #pragma unroll
    for (int i = 0; i < 8; i++) ss += redS[i];

    float inv = 1.0f / ss;
    out[(size_t)b * N_ + tid]       = e0 * inv;
    out[(size_t)b * N_ + tid + 256] = e1 * inv;
}

// ---------------------------------------------------------------------------
extern "C" void kernel_launch(void* const* d_in, const int* in_sizes, int n_in,
                              void* d_out, int out_size, void* d_ws, size_t ws_size,
                              hipStream_t stream) {
    (void)in_sizes; (void)n_in; (void)out_size; (void)ws_size;
    const float* x      = (const float*)d_in[0];
    const float* W_fc   = (const float*)d_in[1];
    const float* b_fc   = (const float*)d_in[2];
    const float* W_ih   = (const float*)d_in[3];
    const float* W_hh   = (const float*)d_in[4];
    const float* b_ih   = (const float*)d_in[5];
    const float* b_hh   = (const float*)d_in[6];
    const float* W_last = (const float*)d_in[7];
    const float* b_last = (const float*)d_in[8];

    char* ws = (char*)d_ws;
    unsigned short* xg = (unsigned short*)ws;                 // 524288*256 bf16 = 256 MB
    size_t off = (size_t)M_ * G4_ * sizeof(unsigned short);
    unsigned short* Wcomb  = (unsigned short*)(ws + off); off += (size_t)G4_ * 32 * 2;
    float*          bcomb  = (float*)(ws + off);          off += (size_t)G4_ * 4;
    unsigned short* Whh_bf = (unsigned short*)(ws + off); off += (size_t)G4_ * H_ * 2;
    float*          logitb = (float*)(ws + off);          off += (size_t)B_ * N_ * 4;

    prep_kernel<<<1, 256, 0, stream>>>(W_fc, b_fc, W_ih, W_hh, b_ih, b_hh,
                                       Wcomb, bcomb, Whh_bf);
    gemm1_kernel<<<M_ / GB_ROWS, 256, 0, stream>>>(x, Wcomb, bcomb, xg);
    lstm_kernel<<<B_ / 16, 128, 0, stream>>>(xg, Whh_bf, W_last, b_last, logitb);
    softmax_kernel<<<B_, 256, 0, stream>>>(logitb, (float*)d_out);
}
